// Attention_1795296330059
// MI455X (gfx1250) — matrix operand order repaired and checked by reading, not compiled, not run on hardware
//
#include <hip/hip_runtime.h>
#include <hip/hip_bf16.h>

// ---------- types ----------
typedef __bf16 bf16x8  __attribute__((ext_vector_type(8)));
typedef __bf16 v16bf   __attribute__((ext_vector_type(16)));
typedef float  v8f     __attribute__((ext_vector_type(8)));

#define SCALE 0.125f  // 64^-0.5

// Assemble a 16x32 bf16 A-fragment slice for this lane:
// lanes 0-15 hold K = {0..15}, lanes 16-31 hold K = {16..31}
// caller passes p already offset by +16*lane_half; we load [p, p+8) and [p+8, p+16).
__device__ __forceinline__ v16bf load_frag_pair(const __bf16* p) {
    bf16x8 lo = *(const bf16x8*)p;
    bf16x8 hi = *(const bf16x8*)(p + 8);
    return __builtin_shufflevector(lo, hi, 0,1,2,3,4,5,6,7,8,9,10,11,12,13,14,15);
}

__device__ __forceinline__ v8f wmma_bf16(v16bf a, v16bf b, v8f c) {
    return __builtin_amdgcn_wmma_f32_16x16x32_bf16(false, a, false, b, (short)0, c, false, false);
}

// ---------- conversion kernels ----------
__global__ void cvt_f32_to_bf16(const float* __restrict__ in, __bf16* __restrict__ out, int n) {
    int i = blockIdx.x * blockDim.x + threadIdx.x;
    int stride = gridDim.x * blockDim.x;
    for (; i < n; i += stride) out[i] = (__bf16)in[i];
}

// in: [K x N] f32 row-major -> out: [N x K] bf16 row-major (transposed)
__global__ void transpose_cvt(const float* __restrict__ in, __bf16* __restrict__ out, int K, int N) {
    int n = blockIdx.x * blockDim.x + threadIdx.x;
    int k = blockIdx.y;
    if (n < N) out[(size_t)n * K + k] = (__bf16)in[(size_t)k * N + n];
}

// ---------- bf16 WMMA GEMM: C[M,N] = A[M,K] * Bt[N,K]^T + bias ----------
// Block: 256 threads = 8 waves arranged 2x4; wave tile 64x32; block tile 128x128.
// mode 0: plain f32 output (proj). mode 1: scatter into per-head Q/K/V^T bf16 (qkv).
__global__ __launch_bounds__(256) void wmma_gemm_bf16(
    const __bf16* __restrict__ A, const __bf16* __restrict__ Bt,
    const float* __restrict__ bias, int N, int K, int mode,
    float* __restrict__ outF, __bf16* __restrict__ q_ws,
    __bf16* __restrict__ k_ws, __bf16* __restrict__ v_t)
{
    const int tid   = threadIdx.x;
    const int wid   = tid >> 5;
    const int lane  = tid & 31;
    const int lrow  = lane & 15;
    const int lhalf = lane >> 4;
    const int rowBase = blockIdx.y * 128 + (wid >> 2) * 64;   // 4 row tiles of 16
    const int colBase = blockIdx.x * 128 + (wid & 3) * 32;    // 2 col tiles of 16

    v8f acc[4][2] = {};

    for (int k0 = 0; k0 < K; k0 += 32) {
        v16bf af[4], bfr[2];
#pragma unroll
        for (int i = 0; i < 4; ++i) {
            const __bf16* ap = A + (size_t)(rowBase + i * 16 + lrow) * K + k0 + 16 * lhalf;
            af[i] = load_frag_pair(ap);
        }
#pragma unroll
        for (int c = 0; c < 2; ++c) {
            const __bf16* bp = Bt + (size_t)(colBase + c * 16 + lrow) * K + k0 + 16 * lhalf;
            bfr[c] = *(const v16bf*)bp;   // 32B contiguous: lane=column, K striped by half
        }
        if (k0 + 32 < K) {  // L2 prefetch hint for next k-step (global_prefetch_b8)
            __builtin_prefetch(A  + (size_t)(rowBase + lrow) * K + k0 + 32, 0, 0);
            __builtin_prefetch(Bt + (size_t)(colBase + lrow) * K + k0 + 32, 0, 0);
        }
#pragma unroll
        for (int i = 0; i < 4; ++i)
#pragma unroll
            for (int c = 0; c < 2; ++c)
                acc[i][c] = wmma_bf16(af[i], bfr[c], acc[i][c]);
    }

    // epilogue: C layout: VGPR v -> row v + 8*lane_half, col = lane&15
#pragma unroll
    for (int i = 0; i < 4; ++i) {
#pragma unroll
        for (int c = 0; c < 2; ++c) {
#pragma unroll
            for (int v = 0; v < 8; ++v) {
                int m = rowBase + i * 16 + v + 8 * lhalf;
                int n = colBase + c * 16 + lrow;
                float val = acc[i][c][v] + bias[n];
                if (mode == 0) {
                    outF[(size_t)m * N + n] = val;
                } else {
                    int b = m >> 10, t = m & 1023;
                    int which = n >> 10, cc = n & 1023, h = cc >> 6, d = cc & 63;
                    size_t bh = (size_t)(b * 16 + h);
                    __bf16 bv = (__bf16)val;
                    if (which == 0)      q_ws[(bh * 1024 + t) * 64 + d] = bv;   // [b][h][tok][d]
                    else if (which == 1) k_ws[(bh * 1024 + t) * 64 + d] = bv;   // [b][h][tok][d]
                    else                 v_t[(bh * 64 + d) * 1024 + t] = bv;    // [b][h][d][tok]
                }
            }
        }
    }
}

// ---------- flash attention (wave-synchronous, per-wave LDS P tile) ----------
// grid (B*H=128, 8); block 256 = 8 waves; wave owns 16 query rows, streams 32 keys/step.
__global__ __launch_bounds__(256) void flash_attn_bf16(
    const __bf16* __restrict__ q_ws, const __bf16* __restrict__ k_ws,
    const __bf16* __restrict__ v_t, __bf16* __restrict__ attn_out)
{
    __shared__ __bf16 p_lds[8][16 * 32];   // per-wave 16x32 bf16 P tile (1 KB each)

    const int tid   = threadIdx.x;
    const int wid   = tid >> 5;
    const int lane  = tid & 31;
    const int lrow  = lane & 15;
    const int lhalf = lane >> 4;
    const int bh = blockIdx.x;            // b*16 + h
    const int b = bh >> 4, h = bh & 15;
    const int qRow0 = blockIdx.y * 128 + wid * 16;

    const __bf16* Q  = q_ws + (size_t)bh * 1024 * 64;
    const __bf16* Km = k_ws + (size_t)bh * 1024 * 64;
    const __bf16* Vt = v_t  + (size_t)bh * 64 * 1024;

    // persistent Q fragments: d in [0,32) and [32,64)
    v16bf qf[2];
#pragma unroll
    for (int kk = 0; kk < 2; ++kk)
        qf[kk] = load_frag_pair(Q + (size_t)(qRow0 + lrow) * 64 + kk * 32 + 16 * lhalf);

    v8f acc[4] = {};
    float mrun[8], lrun[8];
#pragma unroll
    for (int v = 0; v < 8; ++v) { mrun[v] = -__builtin_inff(); lrun[v] = 0.f; }

    for (int j = 0; j < 1024; j += 32) {
        // S tiles: 16 queries x 32 keys = 2 WMMA C-tiles, each reduced over d=64 (2 steps)
        v8f s[2];
#pragma unroll
        for (int t = 0; t < 2; ++t) {
            const __bf16* kb = Km + (size_t)(j + t * 16 + lrow) * 64;  // lane = key column
            v16bf kf0 = *(const v16bf*)(kb + 16 * lhalf);
            v16bf kf1 = *(const v16bf*)(kb + 32 + 16 * lhalf);
            v8f z = {};
            s[t] = wmma_bf16(qf[0], kf0, z);
            s[t] = wmma_bf16(qf[1], kf1, s[t]);
        }

        // online softmax; rows map to (v + 8*lane_half), reductions across 16-lane groups
#pragma unroll
        for (int v = 0; v < 8; ++v) {
            float a0 = s[0][v] * SCALE;
            float a1 = s[1][v] * SCALE;
            float tm = fmaxf(a0, a1);
#pragma unroll
            for (int off = 8; off >= 1; off >>= 1)
                tm = fmaxf(tm, __shfl_xor(tm, off, 32));   // stays within 16-lane group
            float mnew = fmaxf(mrun[v], tm);
            float corr = __expf(mrun[v] - mnew);
            float p0 = __expf(a0 - mnew);
            float p1 = __expf(a1 - mnew);
            float rs = p0 + p1;
#pragma unroll
            for (int off = 8; off >= 1; off >>= 1)
                rs += __shfl_xor(rs, off, 32);
            lrun[v] = lrun[v] * corr + rs;
            mrun[v] = mnew;
#pragma unroll
            for (int dt = 0; dt < 4; ++dt) acc[dt][v] *= corr;
            int r = v + 8 * lhalf;
            p_lds[wid][r * 32 + lrow]      = (__bf16)p0;
            p_lds[wid][r * 32 + 16 + lrow] = (__bf16)p1;
        }
        // wave-private LDS round-trip; DS ops from one wave complete in order (ISA §7.3)
        asm volatile("" ::: "memory");

        // reload P as A-fragment (16 queries x 32 keys)
        v16bf pf = load_frag_pair(&p_lds[wid][lrow * 32 + 16 * lhalf]);

        // O += P * V   (V^T rows contiguous in token)
#pragma unroll
        for (int dt = 0; dt < 4; ++dt) {
            v16bf vf = *(const v16bf*)(Vt + (size_t)(dt * 16 + lrow) * 1024 + j + 16 * lhalf);
            acc[dt] = wmma_bf16(pf, vf, acc[dt]);
        }
    }

    // normalize and store to attn_out [b][tok][h*64+d] (row-major 8192x1024 bf16)
#pragma unroll
    for (int v = 0; v < 8; ++v) {
        float inv = 1.0f / lrun[v];
        int row = qRow0 + v + 8 * lhalf;
#pragma unroll
        for (int dt = 0; dt < 4; ++dt) {
            attn_out[((size_t)b * 1024 + row) * 1024 + h * 64 + dt * 16 + lrow] =
                (__bf16)(acc[dt][v] * inv);
        }
    }
}

// ---------- host launch ----------
extern "C" void kernel_launch(void* const* d_in, const int* in_sizes, int n_in,
                              void* d_out, int out_size, void* d_ws, size_t ws_size,
                              hipStream_t stream) {
    const float* x      = (const float*)d_in[0];   // [8,1024,1024]
    const float* w_qkv  = (const float*)d_in[1];   // [1024,3072]
    const float* b_qkv  = (const float*)d_in[2];   // [3072]
    const float* w_proj = (const float*)d_in[3];   // [1024,1024]
    const float* b_proj = (const float*)d_in[4];   // [1024]
    float* out = (float*)d_out;                    // [8,1024,1024] f32

    char* ws = (char*)d_ws;
    __bf16* xb     = (__bf16*)(ws);                 // 16 MB  [8192,1024]
    __bf16* wqkvt  = (__bf16*)(ws + 16777216);      //  6 MB  [3072,1024]
    __bf16* wprojt = (__bf16*)(ws + 23068672);      //  2 MB  [1024,1024]
    __bf16* q_ws   = (__bf16*)(ws + 25165824);      // 16 MB  [8][16][1024][64]
    __bf16* k_ws   = (__bf16*)(ws + 41943040);      // 16 MB  [8][16][1024][64]
    __bf16* v_t    = (__bf16*)(ws + 58720256);      // 16 MB  [8][16][64][1024]
    __bf16* attn   = (__bf16*)(ws + 75497472);      // 16 MB  [8192,1024]

    cvt_f32_to_bf16<<<2048, 256, 0, stream>>>(x, xb, 8192 * 1024);
    transpose_cvt<<<dim3(12, 1024), 256, 0, stream>>>(w_qkv, wqkvt, 1024, 3072);
    transpose_cvt<<<dim3(4, 1024), 256, 0, stream>>>(w_proj, wprojt, 1024, 1024);

    // QKV: [8192,3072] = xb @ w_qkv, scatter into per-head Q/K/V^T
    wmma_gemm_bf16<<<dim3(24, 64), 256, 0, stream>>>(
        xb, wqkvt, b_qkv, 3072, 1024, /*mode=*/1, nullptr, q_ws, k_ws, v_t);

    // attention per (b,h), 128 query rows per block
    flash_attn_bf16<<<dim3(128, 8), 256, 0, stream>>>(q_ws, k_ws, v_t, attn);

    // projection: [8192,1024] = attn @ w_proj + b_proj -> f32 out
    wmma_gemm_bf16<<<dim3(8, 64), 256, 0, stream>>>(
        attn, wprojt, b_proj, 1024, 1024, /*mode=*/0, out, nullptr, nullptr, nullptr);
}
